// CausalSelfAttention_76149770158531
// MI455X (gfx1250) — matrix operand order repaired and checked by reading, not compiled, not run on hardware
//
#include <hip/hip_runtime.h>

// ---------------------------------------------------------------------------
// Causal self-attention forward, CDNA5 (gfx1250), wave32 + WMMA f16->f32.
//   B=4, T=2048, D=1024, H=16, hd=64.
// Pipeline: qkv GEMM -> flash attention (no LDS, wave-local) -> out GEMM.
// GEMM wave tile 32x64 (2x4 WMMA): 8 wmma per 6 fragment loads.
// ---------------------------------------------------------------------------

typedef __attribute__((ext_vector_type(16))) _Float16 v16h;
typedef __attribute__((ext_vector_type(8)))  float    v8f;

#define NEG_INF (-__builtin_inff())

__device__ __forceinline__ v8f wmma16(v16h a, v16h b, v8f c) {
  // D = A(16x32,f16) * B(32x16,f16) + C(16x16,f32)
  return __builtin_amdgcn_wmma_f32_16x16x32_f16(
      /*neg_a=*/false, a, /*neg_b=*/false, b,
      /*c_mod=*/(short)0, c, /*reuse_a=*/false, /*reuse_b=*/false);
}

// A fragment (16x32 MxK, 16-bit), ISA 7.12.2:
//   lanes 0-15 : row M=row0+l,    K halves {k0+0..7, k0+16..23}
//   lanes 16-31: row M=row0+l-16, K halves {k0+8..15, k0+24..31}
__device__ __forceinline__ v16h fragA_f32(const float* __restrict__ p, int ld,
                                          int row0, int k0) {
  const int lane = threadIdx.x & 31;
  const float* a = p + (size_t)(row0 + (lane & 15)) * ld + (k0 + ((lane >> 4) << 3));
  v16h f;
#pragma unroll
  for (int i = 0; i < 8; ++i) f[i] = (_Float16)a[i];
#pragma unroll
  for (int i = 0; i < 8; ++i) f[8 + i] = (_Float16)a[16 + i];
  return f;
}

__device__ __forceinline__ v16h fragA_f16(const _Float16* __restrict__ p, int ld,
                                          int row0, int k0) {
  const int lane = threadIdx.x & 31;
  const _Float16* a = p + (size_t)(row0 + (lane & 15)) * ld + (k0 + ((lane >> 4) << 3));
  v16h f;
#pragma unroll
  for (int i = 0; i < 8; ++i) f[i] = a[i];
#pragma unroll
  for (int i = 0; i < 8; ++i) f[8 + i] = a[16 + i];
  return f;
}

// B fragment (32x16 KxN, 16-bit). Source stored N-major (row n = K-contiguous):
//   lane l: column N=col0+(l&15), K = k0 + (l>=16 ? 8 : 0) + {0..7, 16..23}
__device__ __forceinline__ v16h fragB_f32(const float* __restrict__ p, int ld,
                                          int col0, int k0) {
  const int lane = threadIdx.x & 31;
  const float* a = p + (size_t)(col0 + (lane & 15)) * ld + (k0 + ((lane >> 4) << 3));
  v16h f;
#pragma unroll
  for (int i = 0; i < 8; ++i) f[i] = (_Float16)a[i];
#pragma unroll
  for (int i = 0; i < 8; ++i) f[8 + i] = (_Float16)a[16 + i];
  return f;
}

__device__ __forceinline__ v16h fragB_f16(const _Float16* __restrict__ p, int ld,
                                          int col0, int k0) {
  const int lane = threadIdx.x & 31;
  const _Float16* a = p + (size_t)(col0 + (lane & 15)) * ld + (k0 + ((lane >> 4) << 3));
  v16h f;
#pragma unroll
  for (int i = 0; i < 8; ++i) f[i] = a[i];
#pragma unroll
  for (int i = 0; i < 8; ++i) f[8 + i] = a[16 + i];
  return f;
}

// ---------------------------------------------------------------------------
// Kernel 1: qkv = x @ w_qkv^T + b_qkv, routed into f16 workspace:
//   Q[bh][t][d], K[bh][t][d] row-major;  V transposed: Vt[bh][d][t].
// Block 256 thr = 8 waves (4x2), block tile 128x128, wave tile 32x64.
// ---------------------------------------------------------------------------
__global__ __launch_bounds__(256) void attn_qkv_gemm(
    const float* __restrict__ x, const float* __restrict__ w,
    const float* __restrict__ bias,
    _Float16* __restrict__ Qh, _Float16* __restrict__ Kh,
    _Float16* __restrict__ Vt) {
  const int wave = threadIdx.x >> 5;
  const int lane = threadIdx.x & 31;
  const int m0 = blockIdx.x * 128 + (wave >> 1) * 32;
  const int n0 = blockIdx.y * 128 + (wave & 1) * 64;

  v8f acc[2][4] = {};
  for (int k0 = 0; k0 < 1024; k0 += 32) {
    if (k0 + 32 < 1024) {  // WGP-scope global_prefetch_b8 of next A k-tile
      __builtin_prefetch(x + (size_t)(m0 + (lane & 15)) * 1024 + k0 + 32, 0, 3);
    }
    v16h a0 = fragA_f32(x, 1024, m0, k0);
    v16h a1 = fragA_f32(x, 1024, m0 + 16, k0);
#pragma unroll
    for (int in = 0; in < 4; ++in) {
      v16h b = fragB_f32(w, 1024, n0 + in * 16, k0);
      acc[0][in] = wmma16(a0, b, acc[0][in]);
      acc[1][in] = wmma16(a1, b, acc[1][in]);
    }
  }

  const int mhi = (lane >> 4) << 3;
#pragma unroll
  for (int in = 0; in < 4; ++in) {
    const int n = n0 + in * 16 + (lane & 15);
    const float bv = bias[n];
    const int which = n >> 10;       // 0=Q 1=K 2=V (reshape order (3,H,hd))
    const int h = (n >> 6) & 15;
    const int d = n & 63;
#pragma unroll
    for (int im = 0; im < 2; ++im) {
#pragma unroll
      for (int j = 0; j < 8; ++j) {
        const int m = m0 + im * 16 + j + mhi;     // global row = b*2048 + t
        const int b = m >> 11;
        const int t = m & 2047;
        const size_t bh = (size_t)((b << 4) | h);
        const _Float16 val = (_Float16)(acc[im][in][j] + bv);
        if (which == 0)      Qh[(bh * 2048 + t) * 64 + d] = val;
        else if (which == 1) Kh[(bh * 2048 + t) * 64 + d] = val;
        else                 Vt[(bh * 64 + d) * 2048 + t] = val;
      }
    }
  }
}

// ---------------------------------------------------------------------------
// Kernel 2: causal flash attention. grid=(64 bh, 32 qtiles), block=128 (4
// waves), each wave owns 16 queries, fully wave-local (no LDS/barriers).
// S^T = K_tile(16x64) x Q^T(64x16) via WMMA; its C-layout (lane=query,
// VGPR=key) IS the A-fragment layout of P for the 16x32 P@V WMMA.
// ---------------------------------------------------------------------------
__global__ __launch_bounds__(128) void attn_flash(
    const _Float16* __restrict__ Qh, const _Float16* __restrict__ Kh,
    const _Float16* __restrict__ Vt, _Float16* __restrict__ O16) {
  const int bh = blockIdx.x;
  const int wave = threadIdx.x >> 5;
  const int lane = threadIdx.x & 31;
  const int q0 = blockIdx.y * 64 + wave * 16;

  const _Float16* Q = Qh + (size_t)bh * 2048 * 64;
  const _Float16* K = Kh + (size_t)bh * 2048 * 64;
  const _Float16* V = Vt + (size_t)bh * 64 * 2048;

  // Q^T as B fragments (dims 0..31, 32..63), resident for whole kb loop
  const v16h qb0 = fragB_f16(Q, 64, q0, 0);
  const v16h qb1 = fragB_f16(Q, 64, q0, 32);

  v8f o[4] = {};                 // O tile: 16 queries x 64 dims (4 N-tiles)
  float mrow = NEG_INF;          // online-softmax stats for query q0+(lane&15)
  float lrow = 0.f;
  const int qg = q0 + (lane & 15);
  const int khi = (lane >> 4) << 3;
  const int bsrc = (lane & 16) >> 1;   // shfl source offset for query j vs j+8
  const float scale = 0.125f;          // hd^-0.5 = 1/8

  for (int kb = 0; kb <= q0 + 15; kb += 32) {   // causal: only live key blocks
    // S^T tiles: rows = keys kb..kb+31, cols = queries q0..q0+15
    v8f s0 = {}, s1 = {};
    s0 = wmma16(fragA_f16(K, 64, kb, 0), qb0, s0);
    s0 = wmma16(fragA_f16(K, 64, kb, 32), qb1, s0);
    s1 = wmma16(fragA_f16(K, 64, kb + 16, 0), qb0, s1);
    s1 = wmma16(fragA_f16(K, 64, kb + 16, 32), qb1, s1);

    // scale + causal mask + block max (element (key kb+j+khi[+16], query qg))
    float p0[8], p1[8];
    float bm = NEG_INF;
#pragma unroll
    for (int j = 0; j < 8; ++j) {
      const int kk = kb + j + khi;
      p0[j] = (kk <= qg) ? s0[j] * scale : NEG_INF;
      p1[j] = (kk + 16 <= qg) ? s1[j] * scale : NEG_INF;
      bm = fmaxf(bm, fmaxf(p0[j], p1[j]));
    }
    bm = fmaxf(bm, __shfl_xor(bm, 16, 32));     // combine lane pair (wave32)
    const float mnew = fmaxf(mrow, bm);
    const float alpha = __expf(mrow - mnew);    // first block: exp(-inf)=0

    // exponentiate + pack P directly into the 16x32 A-fragment (pure per-lane)
    float rsum = 0.f;
    v16h pf;
#pragma unroll
    for (int j = 0; j < 8; ++j) {
      const float e0 = __expf(p0[j] - mnew);
      const float e1 = __expf(p1[j] - mnew);
      rsum += e0 + e1;
      pf[j] = (_Float16)e0;      // keys kb+{0..7 | 8..15}
      pf[8 + j] = (_Float16)e1;  // keys kb+{16..23 | 24..31}
    }
    rsum += __shfl_xor(rsum, 16, 32);
    lrow = lrow * alpha + rsum;
    mrow = mnew;

    // rescale O rows: alpha for query-rel (j + 8*(lane>=16)) lives in that lane
#pragma unroll
    for (int j = 0; j < 8; ++j) {
      const float aj = __shfl(alpha, j + bsrc, 32);
      o[0][j] *= aj; o[1][j] *= aj; o[2][j] *= aj; o[3][j] *= aj;
    }

    // O += P @ V ; Vt is (dim, t) so B-fragments are contiguous per lane
    o[0] = wmma16(pf, fragB_f16(V, 2048, 0, kb), o[0]);
    o[1] = wmma16(pf, fragB_f16(V, 2048, 16, kb), o[1]);
    o[2] = wmma16(pf, fragB_f16(V, 2048, 32, kb), o[2]);
    o[3] = wmma16(pf, fragB_f16(V, 2048, 48, kb), o[3]);
  }

  // finalize: O /= l ; store f16 rows into (B*T, 1024) with col = h*64+d
  const float linv = 1.0f / lrow;
  const int b = bh >> 4, h = bh & 15;
#pragma unroll
  for (int j = 0; j < 8; ++j) {
    const float lj = __shfl(linv, j + bsrc, 32);
    const int t = q0 + j + khi;
    _Float16* row = O16 + (size_t)(b * 2048 + t) * 1024 + h * 64 + (lane & 15);
#pragma unroll
    for (int i = 0; i < 4; ++i) row[16 * i] = (_Float16)(o[i][j] * lj);
  }
}

// ---------------------------------------------------------------------------
// Kernel 3: out = O16 @ w_out^T + b_out  (fp32 result to d_out)
// Block 256 thr = 8 waves (4x2), block tile 128x128, wave tile 32x64.
// ---------------------------------------------------------------------------
__global__ __launch_bounds__(256) void attn_out_gemm(
    const _Float16* __restrict__ A, const float* __restrict__ w,
    const float* __restrict__ bias, float* __restrict__ out) {
  const int wave = threadIdx.x >> 5;
  const int lane = threadIdx.x & 31;
  const int m0 = blockIdx.x * 128 + (wave >> 1) * 32;
  const int n0 = blockIdx.y * 128 + (wave & 1) * 64;

  v8f acc[2][4] = {};
  for (int k0 = 0; k0 < 1024; k0 += 32) {
    if (k0 + 32 < 1024) {
      __builtin_prefetch(A + (size_t)(m0 + (lane & 15)) * 1024 + k0 + 32, 0, 3);
    }
    v16h a0 = fragA_f16(A, 1024, m0, k0);
    v16h a1 = fragA_f16(A, 1024, m0 + 16, k0);
#pragma unroll
    for (int in = 0; in < 4; ++in) {
      v16h b = fragB_f32(w, 1024, n0 + in * 16, k0);
      acc[0][in] = wmma16(a0, b, acc[0][in]);
      acc[1][in] = wmma16(a1, b, acc[1][in]);
    }
  }

  const int mhi = (lane >> 4) << 3;
#pragma unroll
  for (int in = 0; in < 4; ++in) {
    const int n = n0 + in * 16 + (lane & 15);
    const float bv = bias[n];
#pragma unroll
    for (int im = 0; im < 2; ++im) {
#pragma unroll
      for (int j = 0; j < 8; ++j) {
        const int m = m0 + im * 16 + j + mhi;
        out[(size_t)m * 1024 + n] = acc[im][in][j] + bv;
      }
    }
  }
}

// ---------------------------------------------------------------------------
extern "C" void kernel_launch(void* const* d_in, const int* in_sizes, int n_in,
                              void* d_out, int out_size, void* d_ws,
                              size_t ws_size, hipStream_t stream) {
  (void)in_sizes; (void)n_in; (void)out_size; (void)ws_size;

  const float* x     = (const float*)d_in[0];
  // d_in[1] = attn_mask (int32 tril) — causal mask computed analytically
  const float* w_qkv = (const float*)d_in[2];
  const float* b_qkv = (const float*)d_in[3];
  const float* w_out = (const float*)d_in[4];
  const float* b_out = (const float*)d_in[5];

  // f16 workspace: 4 buffers x 8,388,608 elements (16 MB each) = 64 MB
  const size_t SEG = (size_t)64 * 2048 * 64;  // = B*H*T*hd
  _Float16* Qh  = (_Float16*)d_ws;
  _Float16* Kh  = Qh + SEG;
  _Float16* Vt  = Kh + SEG;
  _Float16* O16 = Vt + SEG;

  // qkv: M=8192 (128/block), N=3072 (128/block)
  attn_qkv_gemm<<<dim3(64, 24), 256, 0, stream>>>(x, w_qkv, b_qkv, Qh, Kh, Vt);
  // attention: 64 (b,h) pairs x 32 query-tiles of 64
  attn_flash<<<dim3(64, 32), 128, 0, stream>>>(Qh, Kh, Vt, O16);
  // out proj: M=8192, N=1024 (128/block)
  attn_out_gemm<<<dim3(64, 8), 256, 0, stream>>>(O16, w_out, b_out,
                                                 (float*)d_out);
}
